// TopKAttention_49993419325459
// MI455X (gfx1250) — matrix operand order repaired and checked by reading, not compiled, not run on hardware
//
#include <hip/hip_runtime.h>
#include <hip/hip_bf16.h>
#include <hip/hip_fp16.h>

typedef __attribute__((ext_vector_type(16))) _Float16 v16h;
typedef __attribute__((ext_vector_type(8)))  float    v8f;

#define L_    4096
#define D_    256
#define P_    64
#define B_    4
#define TOPK  128
#define CAP   32768
#define NTIL  32            // 4096 / 128 tiles per dimension
#define SCALE 0.125f        // P^-0.5
#define MARGIN 0.05f        // threshold slack to absorb f16 rounding
#define KSTRIDE 72          // padded LDS row stride (halves): 144B = 36 words,
                            // row*36 mod 64 distinct for 16 rows -> conflict-free

// ---------------------------------------------------------------------------
// Kernel 1: projections  Q = x@Wq + bq, K = x@Wk + bk
// Writes both f16 (WMMA operands) and f32 (exact re-scoring) copies.
// ---------------------------------------------------------------------------
__global__ void proj_kernel(const float* __restrict__ x,
                            const float* __restrict__ Wq, const float* __restrict__ bq,
                            const float* __restrict__ Wk, const float* __restrict__ bk,
                            _Float16* __restrict__ Qh, _Float16* __restrict__ Kh,
                            float* __restrict__ Qf, float* __restrict__ Kf) {
    __shared__ float xs[16 * D_];
    const int tid = threadIdx.x;
    const size_t rowbase = (size_t)blockIdx.x * 16;          // flat row in [0, B*L)
    const float* xsrc = x + rowbase * D_;
    for (int i = tid; i < 16 * D_; i += 256) xs[i] = xsrc[i];
    __syncthreads();

    const int row = tid >> 4;            // 0..15
    const int p0  = (tid & 15) * 4;      // 0,4,...,60
    float q[4], k[4];
#pragma unroll
    for (int j = 0; j < 4; ++j) { q[j] = bq[p0 + j]; k[j] = bk[p0 + j]; }
    for (int d = 0; d < D_; ++d) {
        const float xv = xs[row * D_ + d];
        const float* wq = Wq + d * P_ + p0;
        const float* wk = Wk + d * P_ + p0;
#pragma unroll
        for (int j = 0; j < 4; ++j) {
            q[j] = fmaf(xv, wq[j], q[j]);
            k[j] = fmaf(xv, wk[j], k[j]);
        }
    }
    const size_t o = (rowbase + row) * P_ + p0;
#pragma unroll
    for (int j = 0; j < 4; ++j) {
        Qh[o + j] = (_Float16)q[j];  Kh[o + j] = (_Float16)k[j];
        Qf[o + j] = q[j];            Kf[o + j] = k[j];
    }
}

// ---------------------------------------------------------------------------
// WMMA fragment loaders.
// A 16x32 f16 layout: lane holds row (lane&15); halves 0-15 = K hi*16..+15
// (hi = lane>>4).  Loaded from global Q.
// B 32x16 f16 layout: lane holds col (lane&15); halves 0-15 = K hi*16..+15.
// Loaded from the LDS-staged K tile (padded rows).
// ---------------------------------------------------------------------------
__device__ __forceinline__ v16h load_afrag(const _Float16* p) {
    v16h r;
#pragma unroll
    for (int i = 0; i < 16; ++i) r[i] = p[i];
    return r;
}
__device__ __forceinline__ v16h load_bfrag(const _Float16* p) {
    v16h r;
#pragma unroll
    for (int i = 0; i < 16; ++i) r[i] = p[i];
    return r;
}

// Stage the first 32 halves of each of the 128 K-tile rows into LDS (padded stride), cooperatively.
__device__ __forceinline__ void stage_ktile(_Float16* kt, const _Float16* __restrict__ Kh,
                                            int b, int tcol, int tid) {
    const _Float16* src = Kh + ((size_t)b * L_ + tcol * 128) * P_;
    // 128 rows x 4 chunks of 16B each = 512 chunks, 256 threads -> 2 each
#pragma unroll
    for (int c = tid; c < 512; c += 256) {
        const int row = c >> 2, part = c & 3;
        const uint4 v = *(const uint4*)(src + (size_t)row * P_ + part * 8);
        *(uint4*)(kt + row * KSTRIDE + part * 8) = v;
    }
}

// Compute a 16x128 strip of the 128x128 score tile for one wave.
// acc[st][v] holds scores[row0 + v + 8*hi][tcol*128 + st*16 + (lane&15)].
__device__ __forceinline__ void compute_tile(const _Float16* __restrict__ Qh,
                                             const _Float16* kt /* LDS */,
                                             int b, int trow,
                                             int wave, int lane, v8f acc[8]) {
    const int r  = lane & 15;
    const int hi = lane >> 4;
    const int row0 = trow * 128 + wave * 16;
    const _Float16* qrow = Qh + ((size_t)b * L_ + row0 + r) * P_;
    const v16h a0 = load_afrag(qrow + hi * 16);         // K-chunk 0..31
    const v16h a1 = load_afrag(qrow + 32 + hi * 16);    // K-chunk 32..63
#pragma unroll
    for (int st = 0; st < 8; ++st) {
        const _Float16* krow = kt + (st * 16 + r) * KSTRIDE;
        const v16h b0 = load_bfrag(krow + hi * 16);
        const v16h b1 = load_bfrag(krow + 32 + hi * 16);
        v8f c = {};
        c = __builtin_amdgcn_wmma_f32_16x16x32_f16(false, a0, false, b0,
                                                   (short)0, c, false, false);
        c = __builtin_amdgcn_wmma_f32_16x16x32_f16(false, a1, false, b1,
                                                   (short)0, c, false, false);
        acc[st] = c;
    }
}

// ---------------------------------------------------------------------------
// Kernel 2: per-tile max (masked, scaled). One block per 128x128 tile.
// ---------------------------------------------------------------------------
__global__ void tilemax_kernel(const _Float16* __restrict__ Qh,
                               const _Float16* __restrict__ Kh,
                               const unsigned char* __restrict__ mask,
                               float* __restrict__ tilemax) {
    __shared__ _Float16 kt[128 * KSTRIDE];
    const int blk  = blockIdx.x;
    const int b    = blk >> 10;
    const int t    = blk & 1023;
    const int trow = t >> 5, tcol = t & 31;
    const int tid  = threadIdx.x;
    const int wave = tid >> 5, lane = tid & 31;

    stage_ktile(kt, Kh, b, tcol, tid);
    __syncthreads();

    v8f acc[8];
    compute_tile(Qh, kt, b, trow, wave, lane, acc);

    const unsigned char* m = mask + (size_t)b * L_;
    const int r = lane & 15, hi = lane >> 4;
    const int rowb = trow * 128 + wave * 16 + hi * 8;
    unsigned rmask = 0;
#pragma unroll
    for (int v = 0; v < 8; ++v) rmask |= (m[rowb + v] ? 1u : 0u) << v;

    float lmax = -INFINITY;
#pragma unroll
    for (int st = 0; st < 8; ++st) {
        const int col = tcol * 128 + st * 16 + r;
        if (m[col]) {
#pragma unroll
            for (int v = 0; v < 8; ++v)
                if (rmask & (1u << v)) lmax = fmaxf(lmax, acc[st][v] * SCALE);
        }
    }
    __shared__ float red[256];
    red[tid] = lmax; __syncthreads();
    for (int s = 128; s > 0; s >>= 1) {
        if (tid < s) red[tid] = fmaxf(red[tid], red[tid + s]);
        __syncthreads();
    }
    if (tid == 0) tilemax[(size_t)b * 1024 + t] = red[0];
}

// ---------------------------------------------------------------------------
// Kernel 3: threshold = 128th-largest tile max (valid lower bound on the
// global 128th-largest score). Also resets candidate counters.
// ---------------------------------------------------------------------------
__global__ void threshold_kernel(const float* __restrict__ tilemax,
                                 float* __restrict__ thresh,
                                 int* __restrict__ cnt) {
    const int b = blockIdx.x, tid = threadIdx.x;
    __shared__ float sv[1024];
    __shared__ float rv[256];
    __shared__ int   ri[256];
    for (int i = tid; i < 1024; i += 256) sv[i] = tilemax[(size_t)b * 1024 + i];
    __syncthreads();
    float last = -INFINITY;
    for (int it = 0; it < TOPK; ++it) {
        float lv = -INFINITY; int li = 0;
        for (int i = tid; i < 1024; i += 256)
            if (sv[i] > lv) { lv = sv[i]; li = i; }
        rv[tid] = lv; ri[tid] = li; __syncthreads();
        for (int s = 128; s > 0; s >>= 1) {
            if (tid < s && rv[tid + s] > rv[tid]) { rv[tid] = rv[tid + s]; ri[tid] = ri[tid + s]; }
            __syncthreads();
        }
        if (tid == 0) { last = rv[0]; sv[ri[0]] = -INFINITY; }
        __syncthreads();
    }
    if (tid == 0) { thresh[b] = last; cnt[b] = 0; }
}

// ---------------------------------------------------------------------------
// Kernel 4: recompute tiles (WMMA from LDS), append candidates >= T - MARGIN.
// ---------------------------------------------------------------------------
__global__ void filter_kernel(const _Float16* __restrict__ Qh,
                              const _Float16* __restrict__ Kh,
                              const unsigned char* __restrict__ mask,
                              const float* __restrict__ thresh,
                              int* __restrict__ cnt,
                              int* __restrict__ cidx) {
    __shared__ _Float16 kt[128 * KSTRIDE];
    const int blk  = blockIdx.x;
    const int b    = blk >> 10;
    const int t    = blk & 1023;
    const int trow = t >> 5, tcol = t & 31;
    const int tid  = threadIdx.x;
    const int wave = tid >> 5, lane = tid & 31;

    stage_ktile(kt, Kh, b, tcol, tid);
    __syncthreads();

    v8f acc[8];
    compute_tile(Qh, kt, b, trow, wave, lane, acc);

    const float T = thresh[b] - MARGIN;
    const unsigned char* m = mask + (size_t)b * L_;
    const int r = lane & 15, hi = lane >> 4;
    const int rowb = trow * 128 + wave * 16 + hi * 8;
    unsigned rmask = 0;
#pragma unroll
    for (int v = 0; v < 8; ++v) rmask |= (m[rowb + v] ? 1u : 0u) << v;

#pragma unroll
    for (int st = 0; st < 8; ++st) {
        const int col = tcol * 128 + st * 16 + r;
        if (!m[col]) continue;
#pragma unroll
        for (int v = 0; v < 8; ++v) {
            if (!(rmask & (1u << v))) continue;
            const float val = acc[st][v] * SCALE;
            if (val >= T) {
                const int pos = atomicAdd(&cnt[b], 1);
                if (pos < CAP)
                    cidx[(size_t)b * CAP + pos] = (rowb + v) * L_ + col;
            }
        }
    }
}

// ---------------------------------------------------------------------------
// Kernel 5: fp32 re-score of candidates, exact top-128 selection, softmax,
// emit indices (int32 bits) + weights.  One block per batch.
// ---------------------------------------------------------------------------
__global__ void select_kernel(const int* __restrict__ cnt,
                              volatile float* cval,
                              const int* __restrict__ cidx,
                              const float* __restrict__ Qf,
                              const float* __restrict__ Kf,
                              float* __restrict__ dout) {
    const int b = blockIdx.x, tid = threadIdx.x;
    int n = cnt[b]; if (n > CAP) n = CAP;
    const size_t base = (size_t)b * CAP;

    // fp32 exact re-score of every candidate
    for (int i = tid; i < n; i += 256) {
        const int idx  = cidx[base + i];
        const int rrow = idx >> 12, ccol = idx & 4095;
        const float* qr = Qf + ((size_t)b * L_ + rrow) * P_;
        const float* kr = Kf + ((size_t)b * L_ + ccol) * P_;
        float s = 0.f;
#pragma unroll 8
        for (int p = 0; p < P_; ++p) s = fmaf(qr[p], kr[p], s);
        cval[base + i] = s * SCALE;
    }
    __threadfence();
    __syncthreads();

    __shared__ float rv[256];
    __shared__ int   ri[256];
    __shared__ float selv[TOPK];
    __shared__ int   seli[TOPK];

    for (int it = 0; it < TOPK; ++it) {
        float lv = -INFINITY; int li = -1;
        for (int i = tid; i < n; i += 256) {
            const float v = cval[base + i];
            if (v > lv) { lv = v; li = i; }
        }
        rv[tid] = lv; ri[tid] = li; __syncthreads();
        for (int s = 128; s > 0; s >>= 1) {
            if (tid < s && rv[tid + s] > rv[tid]) { rv[tid] = rv[tid + s]; ri[tid] = ri[tid + s]; }
            __syncthreads();
        }
        if (tid == 0) {
            selv[it] = rv[0];
            seli[it] = (ri[0] >= 0) ? cidx[base + ri[0]] : 0;
            if (ri[0] >= 0) cval[base + ri[0]] = -INFINITY;
        }
        __threadfence();
        __syncthreads();
    }

    // softmax over the 128 selected values (selv[0] is the max)
    __shared__ float ssum[256];
    float e = 0.f;
    if (tid < TOPK)
        e = (selv[tid] > -INFINITY) ? __expf(selv[tid] - selv[0]) : 0.f;
    ssum[tid] = (tid < TOPK) ? e : 0.f; __syncthreads();
    for (int s = 128; s > 0; s >>= 1) {
        if (tid < s) ssum[tid] += ssum[tid + s];
        __syncthreads();
    }
    const float total = ssum[0];

    if (tid < TOPK) {
        const int idx  = seli[tid];
        int* oi = (int*)dout;                       // [B,128,2] int32 bits
        oi[(b * TOPK + tid) * 2]     = idx >> 12;   // row
        oi[(b * TOPK + tid) * 2 + 1] = idx & 4095;  // col
        float* ow = dout + B_ * TOPK * 2;           // [B,128] weights
        ow[b * TOPK + tid] = e / total;
    }
}

// ---------------------------------------------------------------------------
extern "C" void kernel_launch(void* const* d_in, const int* in_sizes, int n_in,
                              void* d_out, int out_size, void* d_ws, size_t ws_size,
                              hipStream_t stream) {
    (void)in_sizes; (void)n_in; (void)out_size; (void)ws_size;

    const float*         x    = (const float*)d_in[0];
    const unsigned char* mask = (const unsigned char*)d_in[1];  // jnp bool -> 1B
    const float*         Wq   = (const float*)d_in[2];
    const float*         bq   = (const float*)d_in[3];
    const float*         Wk   = (const float*)d_in[4];
    const float*         bk   = (const float*)d_in[5];

    char* ws = (char*)d_ws;
    size_t off = 0;
    auto alloc = [&](size_t bytes) -> char* {
        char* p = ws + off;
        off = (off + bytes + 255) & ~(size_t)255;
        return p;
    };
    _Float16* Qh      = (_Float16*)alloc((size_t)B_ * L_ * P_ * sizeof(_Float16));
    _Float16* Kh      = (_Float16*)alloc((size_t)B_ * L_ * P_ * sizeof(_Float16));
    float*    Qf      = (float*)   alloc((size_t)B_ * L_ * P_ * sizeof(float));
    float*    Kf      = (float*)   alloc((size_t)B_ * L_ * P_ * sizeof(float));
    float*    tilemax = (float*)   alloc((size_t)B_ * NTIL * NTIL * sizeof(float));
    float*    thresh  = (float*)   alloc((size_t)B_ * sizeof(float));
    int*      ccnt    = (int*)     alloc((size_t)B_ * sizeof(int));
    float*    cval    = (float*)   alloc((size_t)B_ * CAP * sizeof(float));
    int*      cidx    = (int*)     alloc((size_t)B_ * CAP * sizeof(int));

    proj_kernel<<<(B_ * L_) / 16, 256, 0, stream>>>(x, Wq, bq, Wk, bk, Qh, Kh, Qf, Kf);
    tilemax_kernel<<<B_ * NTIL * NTIL, 256, 0, stream>>>(Qh, Kh, mask, tilemax);
    threshold_kernel<<<B_, 256, 0, stream>>>(tilemax, thresh, ccnt);
    filter_kernel<<<B_ * NTIL * NTIL, 256, 0, stream>>>(Qh, Kh, mask, thresh, ccnt, cidx);
    select_kernel<<<B_, 256, 0, stream>>>(ccnt, cval, cidx, Qf, Kf, (float*)d_out);
}
